// AttentionHead_63831803953509
// MI455X (gfx1250) — compile-verified, matrix operand order checked by reading, not run on hardware
//
#include <hip/hip_runtime.h>

typedef __attribute__((ext_vector_type(16))) _Float16 v16h;
typedef __attribute__((ext_vector_type(8)))  _Float16 v8h;
typedef __attribute__((ext_vector_type(8)))  float    v8f;
typedef __attribute__((ext_vector_type(4)))  float    v4f;

#define BATCH 8
#define SEQ   2048
#define EMB   1024
#define HEAD  64

#define WMMA_F16(A, B, C) \
  __builtin_amdgcn_wmma_f32_16x16x32_f16(false, (A), false, (B), (short)0, (C), false, false)

// 16x32 16-bit fragment, contiguous in k. CDNA5 layout:
//   lanes 0-15:  m=lane,    k in {0..7, 16..23}
//   lanes 16-31: m=lane-16, k in {8..15, 24..31}
// element(m,k) = base[m*mstride + k]. Two 16-byte loads per lane.
static __device__ __forceinline__ v16h frag_f16_contig(const _Float16* base, int mstride) {
  const int lane = threadIdx.x & 31;
  const int m    = lane & 15;
  const int koff = (lane < 16) ? 0 : 8;
  const _Float16* p = base + (size_t)m * mstride + koff;
  v8h g0 = *(const v8h*)(p);
  v8h g1 = *(const v8h*)(p + 16);
  return __builtin_shufflevector(g0, g1, 0, 1, 2, 3, 4, 5, 6, 7, 8, 9, 10, 11, 12, 13, 14, 15);
}

// Same fragment from fp32 source (converted to f16): four 16-byte loads per lane.
static __device__ __forceinline__ v16h frag_f32_contig(const float* base, int mstride) {
  const int lane = threadIdx.x & 31;
  const int m    = lane & 15;
  const int koff = (lane < 16) ? 0 : 8;
  const float* p = base + (size_t)m * mstride + koff;
  v4f a0 = *(const v4f*)(p);
  v4f a1 = *(const v4f*)(p + 4);
  v4f b0 = *(const v4f*)(p + 16);
  v4f b1 = *(const v4f*)(p + 20);
  v16h r;
#pragma unroll
  for (int i = 0; i < 4; ++i) {
    r[i]      = (_Float16)a0[i];
    r[4 + i]  = (_Float16)a1[i];
    r[8 + i]  = (_Float16)b0[i];
    r[12 + i] = (_Float16)b1[i];
  }
  return r;
}

// ---------------------------------------------------------------------------
// Kernel 0: convert the three weight matrices to f16, transposed: Wt[n][k].
// Runs once over 3*64K elements; makes QKV B-fragment loads contiguous b128.
// ---------------------------------------------------------------------------
__global__ __launch_bounds__(256)
void wt_convert_kernel(const float* __restrict__ Wq,
                       const float* __restrict__ Wk,
                       const float* __restrict__ Wv,
                       _Float16* __restrict__ Wt /* 3 x [HEAD][EMB] */) {
  const int tid  = blockIdx.x * 256 + threadIdx.x;  // 0 .. 3*HEAD*EMB-1
  const int midx = tid / (EMB * HEAD);
  const int rem  = tid % (EMB * HEAD);
  const int n    = rem / EMB;
  const int k    = rem % EMB;
  const float* W = (midx == 0) ? Wq : (midx == 1) ? Wk : Wv;
  Wt[tid] = (_Float16)W[(size_t)k * HEAD + n];
}

// ---------------------------------------------------------------------------
// Kernel 1: QKV projection.  X [B*T, C] fp32  x  Wt f16 -> q,k f16 [B*T, H],
// v f16 stored TRANSPOSED as vt [B][H][T] (16B store per lane).
// grid.x = B*T/16; 4 waves/block, wave w owns N-tile w of H=64.
// ---------------------------------------------------------------------------
__global__ __launch_bounds__(128)
void qkv_proj_kernel(const float* __restrict__ X,
                     const _Float16* __restrict__ Wt,
                     _Float16* __restrict__ qo,
                     _Float16* __restrict__ ko,
                     _Float16* __restrict__ vt) {
  const int wave = threadIdx.x >> 5;
  const int lane = threadIdx.x & 31;
  const int n0   = wave * 16;
  const size_t row0 = (size_t)blockIdx.x * 16;
  const float* Xt = X + row0 * EMB;

  const _Float16* Wtq = Wt + (size_t)0 * EMB * HEAD + (size_t)n0 * EMB;
  const _Float16* Wtk = Wt + (size_t)1 * EMB * HEAD + (size_t)n0 * EMB;
  const _Float16* Wtv = Wt + (size_t)2 * EMB * HEAD + (size_t)n0 * EMB;

  v8f cq = {}, ck = {}, cv = {};
  for (int kk = 0; kk < EMB; kk += 32) {
    v16h a  = frag_f32_contig(Xt + kk, EMB);       // A: 16 rows of X
    v16h bq = frag_f16_contig(Wtq + kk, EMB);      // B: element(k,n)=Wt[n][kk+k]
    v16h bk = frag_f16_contig(Wtk + kk, EMB);
    v16h bv = frag_f16_contig(Wtv + kk, EMB);
    cq = WMMA_F16(a, bq, cq);
    ck = WMMA_F16(a, bk, ck);
    cv = WMMA_F16(a, bv, cv);
  }

  // C/D layout: lane n (col), vgpr r -> row r (lanes 0-15) / r+8 (lanes 16-31)
  const int col   = lane & 15;
  const int rbase = (lane < 16) ? 0 : 8;

  // q, k stored [token][head]
#pragma unroll
  for (int r = 0; r < 8; ++r) {
    const size_t idx = (row0 + rbase + r) * HEAD + n0 + col;
    qo[idx] = (_Float16)cq[r];
    ko[idx] = (_Float16)ck[r];
  }

  // v stored transposed [b][h][t]; each lane holds 8 consecutive t for fixed h
  {
    const int bidx = (int)(row0 / SEQ);
    const int t0   = (int)(row0 % SEQ);
    v8h pv;
#pragma unroll
    for (int r = 0; r < 8; ++r) pv[r] = (_Float16)cv[r];
    _Float16* vp = vt + ((size_t)bidx * HEAD + n0 + col) * SEQ + t0 + rbase;
    *(v8h*)vp = pv;
  }
}

// ---------------------------------------------------------------------------
// Kernel 2: causal flash attention.  One 16-query tile per wave.
//   S^T = K . Q^T  -> C layout lane == query; online softmax per lane
//   (+ one shfl_xor(16) pairing lanes n and n+16); P repacks in-lane into
//   the B fragment for  O^T += V^T . P  (V^T loads contiguous from vt).
// grid = (T/64, B), 4 waves/block.
// ---------------------------------------------------------------------------
__global__ __launch_bounds__(128)
void flash_attn_kernel(const _Float16* __restrict__ qb,
                       const _Float16* __restrict__ kb,
                       const _Float16* __restrict__ vt,
                       float* __restrict__ out) {
  const int wave = threadIdx.x >> 5;
  const int lane = threadIdx.x & 31;
  const int b    = blockIdx.y;
  const int q0   = (blockIdx.x * 4 + wave) * 16;

  const _Float16* qp  = qb + (size_t)b * SEQ * HEAD;
  const _Float16* kp  = kb + (size_t)b * SEQ * HEAD;
  const _Float16* vtb = vt + (size_t)b * HEAD * SEQ;

  // Q^T as B operand: element(k=h, n=q) = qp[(q0+n)*HEAD + h0 + k]
  const v16h bq0 = frag_f16_contig(qp + (size_t)q0 * HEAD + 0,  HEAD);
  const v16h bq1 = frag_f16_contig(qp + (size_t)q0 * HEAD + 32, HEAD);

  v8f o0 = {}, o1 = {}, o2 = {}, o3 = {};
  float mrun = -1e30f, lrun = 0.0f;

  const int   col   = lane & 15;
  const int   rbase = (lane < 16) ? 0 : 8;
  const int   qidx  = q0 + col;
  const float scale = 0.03125f;  // C**-0.5 = 1024**-0.5

  const int send = q0 + 16;
  for (int s0 = 0; s0 < send; s0 += 32) {
    if (s0 + 32 < send) {  // uniform: prefetch next chunk of K and V^T
      __builtin_prefetch(kp + (size_t)(s0 + 32) * HEAD, 0, 0);
      __builtin_prefetch(vtb + (size_t)(lane & 48) * SEQ + s0 + 32, 0, 0);
    }

    // S^T tiles: keys [s0,s0+16) and [s0+16,s0+32), K-dim = H = 64
    v8f st0 = {}, st1 = {};
    {
      v16h a0 = frag_f16_contig(kp + (size_t)s0 * HEAD + 0,  HEAD);
      st0 = WMMA_F16(a0, bq0, st0);
      v16h a1 = frag_f16_contig(kp + (size_t)s0 * HEAD + 32, HEAD);
      st0 = WMMA_F16(a1, bq1, st0);
      v16h a2 = frag_f16_contig(kp + (size_t)(s0 + 16) * HEAD + 0,  HEAD);
      st1 = WMMA_F16(a2, bq0, st1);
      v16h a3 = frag_f16_contig(kp + (size_t)(s0 + 16) * HEAD + 32, HEAD);
      st1 = WMMA_F16(a3, bq1, st1);
    }

    // scale + causal mask (per-lane selects; EXEC stays all-ones)
    float sc0[8], sc1[8];
    float cmax = -1e30f;
#pragma unroll
    for (int r = 0; r < 8; ++r) {
      const int k0i = s0 + rbase + r;
      const int k1i = s0 + 16 + rbase + r;
      const float x0 = (k0i <= qidx) ? st0[r] * scale : -1e30f;
      const float x1 = (k1i <= qidx) ? st1[r] * scale : -1e30f;
      sc0[r] = x0; sc1[r] = x1;
      cmax = fmaxf(cmax, fmaxf(x0, x1));
    }
    cmax = fmaxf(cmax, __shfl_xor(cmax, 16));
    const float mnew  = fmaxf(mrun, cmax);
    const float alpha = __expf(mrun - mnew);

    // exponentiate, repack in-lane into the B fragment for P.V
    v16h pb = {};
    float psum = 0.0f;
#pragma unroll
    for (int r = 0; r < 8; ++r) {
      const float p0 = __expf(sc0[r] - mnew);
      const float p1 = __expf(sc1[r] - mnew);
      psum += p0 + p1;
      pb[r]     = (_Float16)p0;
      pb[8 + r] = (_Float16)p1;
    }
    psum += __shfl_xor(psum, 16);
    lrun = lrun * alpha + psum;
    mrun = mnew;

#pragma unroll
    for (int r = 0; r < 8; ++r) {
      o0[r] *= alpha; o1[r] *= alpha; o2[r] *= alpha; o3[r] *= alpha;
    }

    // O^T += V^T . P : A = V^T, element(m=h, k=key) = vt[h0+m][s0+k] (contiguous k)
    v16h av0 = frag_f16_contig(vtb + (size_t)( 0) * SEQ + s0, SEQ);
    o0 = WMMA_F16(av0, pb, o0);
    v16h av1 = frag_f16_contig(vtb + (size_t)(16) * SEQ + s0, SEQ);
    o1 = WMMA_F16(av1, pb, o1);
    v16h av2 = frag_f16_contig(vtb + (size_t)(32) * SEQ + s0, SEQ);
    o2 = WMMA_F16(av2, pb, o2);
    v16h av3 = frag_f16_contig(vtb + (size_t)(48) * SEQ + s0, SEQ);
    o3 = WMMA_F16(av3, pb, o3);
  }

  const float inv_l = 1.0f / lrun;
  float* op = out + ((size_t)b * SEQ + qidx) * HEAD + rbase;

  v4f w0, w1;
#pragma unroll
  for (int r = 0; r < 4; ++r) { w0[r] = o0[r] * inv_l; w1[r] = o0[4 + r] * inv_l; }
  *(v4f*)(op + 0) = w0; *(v4f*)(op + 4) = w1;
#pragma unroll
  for (int r = 0; r < 4; ++r) { w0[r] = o1[r] * inv_l; w1[r] = o1[4 + r] * inv_l; }
  *(v4f*)(op + 16) = w0; *(v4f*)(op + 20) = w1;
#pragma unroll
  for (int r = 0; r < 4; ++r) { w0[r] = o2[r] * inv_l; w1[r] = o2[4 + r] * inv_l; }
  *(v4f*)(op + 32) = w0; *(v4f*)(op + 36) = w1;
#pragma unroll
  for (int r = 0; r < 4; ++r) { w0[r] = o3[r] * inv_l; w1[r] = o3[4 + r] * inv_l; }
  *(v4f*)(op + 48) = w0; *(v4f*)(op + 52) = w1;
}

extern "C" void kernel_launch(void* const* d_in, const int* in_sizes, int n_in,
                              void* d_out, int out_size, void* d_ws, size_t ws_size,
                              hipStream_t stream) {
  (void)in_sizes; (void)n_in; (void)out_size; (void)ws_size;
  const float* data = (const float*)d_in[0];
  const float* Wq   = (const float*)d_in[1];
  const float* Wk   = (const float*)d_in[2];
  const float* Wv   = (const float*)d_in[3];
  float* out = (float*)d_out;

  const size_t nqkv = (size_t)BATCH * SEQ * HEAD;  // 1M elems -> 2 MB per f16 buffer
  _Float16* qbuf = (_Float16*)d_ws;
  _Float16* kbuf = qbuf + nqkv;
  _Float16* vtbf = kbuf + nqkv;                    // transposed V [B][H][T]
  _Float16* wtbf = vtbf + nqkv;                    // 3 x [HEAD][EMB] f16 (384 KB)

  wt_convert_kernel<<<dim3(3 * EMB * HEAD / 256), 256, 0, stream>>>(Wq, Wk, Wv, wtbf);
  qkv_proj_kernel<<<dim3(BATCH * SEQ / 16), 128, 0, stream>>>(data, wtbf, qbuf, kbuf, vtbf);
  flash_attn_kernel<<<dim3(SEQ / 64, BATCH), 128, 0, stream>>>(qbuf, kbuf, vtbf, out);
}
